// RNN_4157528342647
// MI455X (gfx1250) — matrix operand order repaired and checked by reading, not compiled, not run on hardware
//
#include <hip/hip_runtime.h>
#include <hip/hip_bf16.h>
#include <math.h>

// ---------------------------------------------------------------------------
// 2-layer GELU RNN, T=2048 B=128 I=128 H=256.
//  pass 0: convert sequence + weights to f16 once (enables async LDS staging)
//  pass 1: Xi = X @ Wi^T  (parallel WMMA GEMM over T*B rows)
//  pass 2: scan, 8 WGs x 16 independent batch rows, Wh resident in LDS (f16),
//          per step one 16x256 @ 256x256 WMMA GEMM + one barrier.
// LDS tiles staged with GLOBAL_LOAD_ASYNC_TO_LDS_B128 (ASYNCcnt) when the
// builtin is available; plain copies otherwise.
// ---------------------------------------------------------------------------

typedef __attribute__((ext_vector_type(16))) _Float16 v16h;
typedef __attribute__((ext_vector_type(8)))  _Float16 v8h;
typedef __attribute__((ext_vector_type(8)))  float    v8f;
typedef __attribute__((ext_vector_type(4)))  int      v4i;

#define Tn 2048
#define Bn 128
#define In 128
#define Hn 256

#if __has_builtin(__builtin_amdgcn_global_load_async_to_lds_b128)
#define HAS_ASYNC_LDS 1
#endif

union FragH { v16h v; v8h h[2]; };

// ---- A fragment: 16x32 f16 tile, row-major in LDS, row stride ld ----------
__device__ inline v16h load_frag_a(const _Float16* a, int row0, int ld,
                                   int kb, int lane) {
  const int half = lane >> 4;
  const int r    = lane & 15;
  const _Float16* p = a + (size_t)(row0 + r) * ld + kb + half * 8;
  FragH f;
  f.h[0] = *(const v8h*)(p);
  f.h[1] = *(const v8h*)(p + 16);
  return f.v;
}

// ---- B fragment: 32x16 f16 tile of B[k][n] = W[n][k], W row-major ---------
__device__ inline v16h load_frag_b(const _Float16* w, int col0, int ld,
                                   int kb, int lane) {
  const int col = lane & 15;
  const int kh  = lane >> 4;
  const _Float16* p = w + (size_t)(col0 + col) * ld + kb + kh * 8;
  FragH f;
  f.h[0] = *(const v8h*)(p);
  f.h[1] = *(const v8h*)(p + 16);
  return f.v;
}

__device__ inline float gelu_exact(float x) {
  return 0.5f * x * (1.0f + erff(x * 0.70710678118654752f));
}

// ---- async stage: copy nHalf f16 elems global -> LDS, 16B chunks ----------
__device__ inline void stage_lds(_Float16* dst, const _Float16* __restrict__ src,
                                 int nHalf, int tid, int nthreads) {
#ifdef HAS_ASYNC_LDS
  _Float16* s = const_cast<_Float16*>(src);
  for (int i = tid * 8; i < nHalf; i += nthreads * 8) {
    __builtin_amdgcn_global_load_async_to_lds_b128(
        (__attribute__((address_space(1))) v4i*)(s + i),
        (__attribute__((address_space(3))) v4i*)(dst + i),
        0, 0);
  }
#else
  const unsigned int* s32 = (const unsigned int*)src;
  unsigned int* d32 = (unsigned int*)dst;
  for (int i = tid; i < nHalf / 2; i += nthreads) d32[i] = s32[i];
#endif
}

__device__ inline void stage_wait() {
#ifdef HAS_ASYNC_LDS
#if __has_builtin(__builtin_amdgcn_s_wait_asynccnt)
  __builtin_amdgcn_s_wait_asynccnt(0);
#else
  asm volatile("s_wait_asynccnt 0" ::: "memory");
#endif
#endif
}

// ---------------------------------------------------------------------------
// one-shot f32 -> f16 conversion (8 elems/thread, vectorized)
// ---------------------------------------------------------------------------
__global__ void cvt_f16(const float* __restrict__ s, _Float16* __restrict__ d,
                        int n) {
  int i = (blockIdx.x * 256 + threadIdx.x) * 8;
  if (i + 8 <= n) {
    #pragma unroll
    for (int j = 0; j < 8; ++j) d[i + j] = (_Float16)s[i + j];
  }
}

// ---------------------------------------------------------------------------
// Xi = X @ W^T.  X: [Rtotal][K] f16, W: [256][K] f16, Out: [Rtotal][256] f32
// block = 512 (16 waves), grid = Rtotal/64.  Wave w owns column tile w.
// ---------------------------------------------------------------------------
__global__ void xi_gemm(const _Float16* __restrict__ X,
                        const _Float16* __restrict__ W,
                        float* __restrict__ Out, int K) {
  extern __shared__ _Float16 smem[];
  _Float16* sW = smem;               // [256][K]
  _Float16* sX = smem + 256 * K;     // [64][K]

  const int tid  = threadIdx.x;
  const int lane = tid & 31;
  const int wave = tid >> 5;
  const int r0   = blockIdx.x * 64;

  stage_lds(sW, W, 256 * K, tid, 512);
  stage_lds(sX, X + (size_t)r0 * K, 64 * K, tid, 512);
  stage_wait();
  __syncthreads();

  const int n0 = wave * 16;
  const int nk = K >> 5;
  for (int mt = 0; mt < 4; ++mt) {
    v8f acc = {0.f, 0.f, 0.f, 0.f, 0.f, 0.f, 0.f, 0.f};
    for (int kb = 0; kb < nk; ++kb) {
      v16h a = load_frag_a(sX, mt * 16, K, kb * 32, lane);
      v16h b = load_frag_b(sW, n0, K, kb * 32, lane);
      acc = __builtin_amdgcn_wmma_f32_16x16x32_f16(
          false, a, false, b, (short)0, acc, false, false);
    }
    const int n  = n0 + (lane & 15);
    const int mb = 8 * (lane >> 4);
    #pragma unroll
    for (int r = 0; r < 8; ++r) {
      Out[(size_t)(r0 + mt * 16 + mb + r) * 256 + n] = acc[r];
    }
  }
}

// ---------------------------------------------------------------------------
// Sequential scan: grid = 8 (16 batch rows each), block = 512 (16 waves).
// ---------------------------------------------------------------------------
__global__ void rnn_scan(const _Float16* __restrict__ WhH,  // [256][256] f16
                         const float* __restrict__ init,    // [256]
                         const float* __restrict__ Xi,      // [T][B][256] f32
                         float* __restrict__ yF32,          // layer1 out or null
                         _Float16* __restrict__ yF16,       // layer0 out or null
                         float* __restrict__ hFinal) {      // [B][256]
  extern __shared__ _Float16 smem[];
  _Float16* sWh = smem;                        // 256*256
  _Float16* sh0 = smem + 256 * 256;            // 16*256
  _Float16* sh1 = sh0 + 16 * 256;              // 16*256

  const int tid  = threadIdx.x;
  const int lane = tid & 31;
  const int wave = tid >> 5;
  const int wg   = blockIdx.x;
  const int n0   = wave * 16;

  stage_lds(sWh, WhH, 256 * 256, tid, 512);
  for (int i = tid; i < 16 * 256; i += 512) sh0[i] = (_Float16)init[i & 255];
  stage_wait();
  __syncthreads();

  const int n  = n0 + (lane & 15);
  const int mb = 8 * (lane >> 4);

  for (int t = 0; t < Tn; ++t) {
    _Float16* shc = (t & 1) ? sh1 : sh0;
    _Float16* shn = (t & 1) ? sh0 : sh1;
    const float* xi = Xi + ((size_t)t * Bn + wg * 16) * Hn;

    // issue per-lane Xi loads first so they overlap the WMMA chain
    float xiv[8];
    #pragma unroll
    for (int r = 0; r < 8; ++r) xiv[r] = xi[(size_t)(mb + r) * Hn + n];

    v8f acc = {0.f, 0.f, 0.f, 0.f, 0.f, 0.f, 0.f, 0.f};
    #pragma unroll
    for (int kb = 0; kb < 8; ++kb) {
      v16h a = load_frag_a(shc, 0, 256, kb * 32, lane);
      v16h b = load_frag_b(sWh, n0, 256, kb * 32, lane);
      acc = __builtin_amdgcn_wmma_f32_16x16x32_f16(
          false, a, false, b, (short)0, acc, false, false);
    }

    if (t + 1 < Tn) {
      const float* xin = Xi + ((size_t)(t + 1) * Bn + wg * 16) * Hn;
      __builtin_prefetch(xin + (size_t)tid * 8, 0, 1);
    }

    #pragma unroll
    for (int r = 0; r < 8; ++r) {
      const int m = mb + r;
      float v = acc[r] + xiv[r];
      v = gelu_exact(v);
      const size_t gidx = ((size_t)t * Bn + wg * 16 + m) * Hn + n;
      if (yF32) yF32[gidx] = v;
      if (yF16) yF16[gidx] = (_Float16)v;
      shn[m * 256 + n] = (_Float16)v;
      if (t == Tn - 1) hFinal[(size_t)(wg * 16 + m) * Hn + n] = v;
    }
    __syncthreads();
  }
}

// ---------------------------------------------------------------------------
extern "C" void kernel_launch(void* const* d_in, const int* in_sizes, int n_in,
                              void* d_out, int out_size, void* d_ws, size_t ws_size,
                              hipStream_t stream) {
  const float* seq  = (const float*)d_in[0];   // [T,B,I]
  const float* init = (const float*)d_in[1];   // [2,1,H]
  const float* Wi0  = (const float*)d_in[2];   // [H,I]
  const float* Wh0  = (const float*)d_in[3];   // [H,H]
  const float* Wi1  = (const float*)d_in[4];   // [H,H]
  const float* Wh1  = (const float*)d_in[5];   // [H,H]

  float* y1   = (float*)d_out;
  float* hFin = y1 + (size_t)Tn * Bn * Hn;

  // workspace layout
  char* ws = (char*)d_ws;
  float*    Xi   = (float*)ws;                                   // 256 MB
  _Float16* y0   = (_Float16*)(ws + (size_t)Tn * Bn * Hn * 4);   // 128 MB
  _Float16* seqH = (_Float16*)(ws + (size_t)Tn * Bn * Hn * 6);   //  64 MB
  _Float16* wWi0 = seqH + (size_t)Tn * Bn * In;
  _Float16* wWh0 = wWi0 + Hn * In;
  _Float16* wWi1 = wWh0 + Hn * Hn;
  _Float16* wWh1 = wWi1 + Hn * Hn;

  const int R = Tn * Bn;

  // pass 0: f32 -> f16 conversions
  cvt_f16<<<(Tn * Bn * In) / 2048, 256, 0, stream>>>(seq, seqH, Tn * Bn * In);
  cvt_f16<<<(Hn * In) / 2048, 256, 0, stream>>>(Wi0, wWi0, Hn * In);
  cvt_f16<<<(Hn * Hn) / 2048, 256, 0, stream>>>(Wh0, wWh0, Hn * Hn);
  cvt_f16<<<(Hn * Hn) / 2048, 256, 0, stream>>>(Wi1, wWi1, Hn * Hn);
  cvt_f16<<<(Hn * Hn) / 2048, 256, 0, stream>>>(Wh1, wWh1, Hn * Hn);

  // layer 0
  {
    size_t shm = (size_t)(256 * In + 64 * In) * sizeof(_Float16);
    xi_gemm<<<R / 64, 512, shm, stream>>>(seqH, wWi0, Xi, In);
  }
  {
    size_t shm = (size_t)(256 * 256 + 2 * 16 * 256) * sizeof(_Float16);
    rnn_scan<<<8, 512, shm, stream>>>(wWh0, init, Xi, nullptr, y0, hFin);
  }
  // layer 1
  {
    size_t shm = (size_t)(256 * Hn + 64 * Hn) * sizeof(_Float16);
    xi_gemm<<<R / 64, 512, shm, stream>>>(y0, wWi1, Xi, Hn);
  }
  {
    size_t shm = (size_t)(256 * 256 + 2 * 16 * 256) * sizeof(_Float16);
    rnn_scan<<<8, 512, shm, stream>>>(wWh1, init + Hn, Xi, y1, nullptr,
                                      hFin + (size_t)Bn * Hn);
  }
}
